// multi_head_attn_5566277615839
// MI455X (gfx1250) — matrix operand order repaired and checked by reading, not compiled, not run on hardware
//
#include <hip/hip_runtime.h>
#include <hip/hip_bf16.h>

// Problem constants (from reference): H=8, F=256, W=96, Hh=96, C=64
#define F_DIM 256
#define SP    9216      // 96*96 spatial positions
#define WW    96
#define HHH   96
#define PW    98        // padded spatial edge (halo of 1 for 3x3 conv)
#define C_DIM 64
#define NH    8
#define AT_WAVES 4
#define KSPLIT  4
#define KEYS_PER_SPLIT (SP / KSPLIT)   // 2304
#define NCHUNK (KEYS_PER_SPLIT / 32)   // 72

typedef __bf16 bf16;
typedef __attribute__((ext_vector_type(16))) __bf16 v16bf;
typedef __attribute__((ext_vector_type(8)))  float  v8f;

struct B128 { unsigned int x, y, z, w; };
union FragU { B128 b[2]; v16bf v; };

// ---- WMMA fragment loaders (CDNA5 ISA §7.12.2, wave32) -------------
// A (16x32 bf16): lane l -> row m=l&15; half=l>>4. VGPR0..3 = K half*8..+7,
// VGPR4..7 = K 16+half*8..+7. Source row-major with K contiguous.
template <typename P>
static __device__ inline v16bf load_a(const P* rowptr, int half) {
  FragU f;
  f.b[0] = *(const B128*)(rowptr + half * 8);
  f.b[1] = *(const B128*)(rowptr + 16 + half * 8);
  return f.v;
}
// B (32x16 bf16): lane l -> col n=l&15; half=l>>4. VGPR0..3 = K half*8..+7,
// VGPR4..7 = K 16+half*8..+7. Source (N,K)-major with K contiguous.
template <typename P>
static __device__ inline v16bf load_b(const P* rowptr, int half) {
  FragU f;
  f.b[0] = *(const B128*)(rowptr + half * 8);
  f.b[1] = *(const B128*)(rowptr + 16 + half * 8);
  return f.v;
}
static __device__ inline v8f wmma_bf16(v16bf a, v16bf b, v8f c) {
  return __builtin_amdgcn_wmma_f32_16x16x32_bf16(false, a, false, b, (short)0, c,
                                                 false, false);
}
// CDNA5 async global->LDS copy (16B per lane), tracked by ASYNCcnt.
static __device__ inline void async_copy16(unsigned lds_off, const void* gptr) {
  asm volatile("global_load_async_to_lds_b128 %0, %1, off"
               :: "v"(lds_off), "v"(gptr) : "memory");
}
static __device__ inline void wait_async0() {
  asm volatile("s_wait_asynccnt 0x0" ::: "memory");
}

// ---- Kernel 1: fold head-sum + affines; emit QsT / KT / padded XT -------
__global__ __launch_bounds__(256) void qk_prep(
    const float* __restrict__ X, const float* __restrict__ WQ_task,
    const float* __restrict__ BQ_task, const float* __restrict__ WK_task,
    const float* __restrict__ BK_task, const float* __restrict__ WQ_tm1,
    const float* __restrict__ WQ_x, const float* __restrict__ BQ,
    const float* __restrict__ WK_x, const float* __restrict__ BK,
    const float* __restrict__ prevQ, bf16* __restrict__ QsT,
    bf16* __restrict__ KT, bf16* __restrict__ XTpad) {
  const int f = blockIdx.x;                       // uniform -> scalar loads
  const int p = blockIdx.y * blockDim.x + threadIdx.x;
  float acoef[NH];
  float bsum = 0.f, csum = 0.f;
#pragma unroll
  for (int h = 0; h < NH; ++h) {
    float wq = WQ_task[h * F_DIM + f];
    acoef[h] = wq * WQ_tm1[h * F_DIM + f];
    bsum += wq;
    csum += wq * BQ[h * F_DIM + f] + BQ_task[h * F_DIM + f];
  }
  const float x = X[f * SP + p];
  float qs = bsum * WQ_x[f] * x + csum;
#pragma unroll
  for (int h = 0; h < NH; ++h) qs += acoef[h] * prevQ[(h * F_DIM + f) * SP + p];
  const float k = WK_task[f] * (WK_x[f] * x + BK[f]) + BK_task[f];
  QsT[p * F_DIM + f] = (bf16)qs;
  KT [p * F_DIM + f] = (bf16)k;
  const int w = p / HHH, h2 = p - w * HHH;
  XTpad[((w + 1) * PW + h2 + 1) * F_DIM + f] = (bf16)x;   // interior of halo
}

// ---- Kernel 2: conv weights (C,F,3,3) f32 -> (tap, c, f) bf16 -----------
__global__ __launch_bounds__(256) void w_prep(const float* __restrict__ Vw,
                                              bf16* __restrict__ Wc) {
  int tid = blockIdx.x * blockDim.x + threadIdx.x;   // (j*64+c)*256+f
  if (tid >= 9 * C_DIM * F_DIM) return;
  int fidx = tid & (F_DIM - 1);
  int t2 = tid >> 8;
  int c = t2 & (C_DIM - 1);
  int j = t2 >> 6;
  Wc[tid] = (bf16)Vw[(c * F_DIM + fidx) * 9 + j];
}

// ---- Kernel 3: 3x3 conv as tap-decomposed WMMA GEMM + fused affine ------
// Padded XT => every tap load is unconditionally in-bounds (no masks).
__global__ __launch_bounds__(128) void conv_v(
    const bf16* __restrict__ XTpad, const bf16* __restrict__ Wc,
    const float* __restrict__ WV_task, const float* __restrict__ BV_task,
    const float* __restrict__ Vbias, bf16* __restrict__ Vbuf) {
  const int lane = threadIdx.x & 31;
  const int wave = threadIdx.x >> 5;
  const int pbase = (blockIdx.x * (blockDim.x >> 5) + wave) * 16;
  const int m = lane & 15, half = lane >> 4;
  const int p = pbase + m;
  const int w = p / HHH, h = p - w * HHH;           // p = w*96 + h
  const int rp = (w + 1) * PW + (h + 1);            // padded row index
  v8f acc0 = {}, acc1 = {}, acc2 = {}, acc3 = {};
#pragma unroll
  for (int j = 0; j < 9; ++j) {
    const int dw = j / 3 - 1, dh = j % 3 - 1;
    const bf16* xr = XTpad + (size_t)(rp + dw * PW + dh) * F_DIM;
    const bf16* wr = Wc + (size_t)(j * C_DIM + m) * F_DIM;
#pragma unroll
    for (int fc = 0; fc < 8; ++fc) {
      v16bf a = load_a(xr + fc * 32, half);
      acc0 = wmma_bf16(a, load_b(wr + 0 * 16 * F_DIM + fc * 32, half), acc0);
      acc1 = wmma_bf16(a, load_b(wr + 1 * 16 * F_DIM + fc * 32, half), acc1);
      acc2 = wmma_bf16(a, load_b(wr + 2 * 16 * F_DIM + fc * 32, half), acc2);
      acc3 = wmma_bf16(a, load_b(wr + 3 * 16 * F_DIM + fc * 32, half), acc3);
    }
  }
  v8f* accs[4] = {&acc0, &acc1, &acc2, &acc3};
#pragma unroll
  for (int ct = 0; ct < 4; ++ct) {
    const int c = ct * 16 + m;
    const float wv = WV_task[c], bv = BV_task[c], cb = Vbias[c];
#pragma unroll
    for (int v = 0; v < 8; ++v) {
      const int pr = pbase + v + 8 * half;
      Vbuf[(size_t)c * SP + pr] = (bf16)(wv * ((*accs[ct])[v] + cb) + bv);
    }
  }
}

// ---- Kernel 4: flash attention, split-K over keys, LDS-staged K/V -------
__global__ __launch_bounds__(AT_WAVES * 32) void attn(
    const bf16* __restrict__ QsT, const bf16* __restrict__ KT,
    const bf16* __restrict__ Vbuf, float* __restrict__ Opart,
    float* __restrict__ Mpart, float* __restrict__ Lpart) {
  __shared__ __align__(16) bf16 Ks[2][32 * F_DIM];   // 16 KB x2
  __shared__ __align__(16) bf16 Vs[2][C_DIM * 32];   // 4 KB x2
  __shared__ __align__(16) bf16 Ps[AT_WAVES][16 * 32];
  const int tid = threadIdx.x;
  const int lane = tid & 31;
  const int wave = tid >> 5;
  const int split = blockIdx.y;
  const int kbeg = split * KEYS_PER_SPLIT;
  const int qbase = (blockIdx.x * AT_WAVES + wave) * 16;
  const int m = lane & 15, half = lane >> 4;
  bf16* lds = Ps[wave];

  // Cooperative async stage of one 32-key chunk of K (32x256) and V (64x32).
  auto stage = [&](int pk_local, int buf) {
    const char* ksrc = (const char*)(KT + (size_t)(kbeg + pk_local) * F_DIM);
    unsigned kd = (unsigned)(size_t)&Ks[buf][0];
#pragma unroll
    for (int i = 0; i < 8; ++i) {
      int idx = tid + i * (AT_WAVES * 32);
      async_copy16(kd + idx * 16, ksrc + idx * 16);
    }
    unsigned vd = (unsigned)(size_t)&Vs[buf][0];
#pragma unroll
    for (int i = 0; i < 2; ++i) {
      int idx = tid + i * (AT_WAVES * 32);     // 0..255: c=idx>>2, quad=idx&3
      const char* vsrc = (const char*)(Vbuf + (size_t)(idx >> 2) * SP + kbeg +
                                       pk_local + (idx & 3) * 8);
      async_copy16(vd + idx * 16, vsrc);
    }
  };

  // Resident A fragments: 16 query rows x 256 features (64 VGPRs)
  v16bf Afrag[8];
  const bf16* qrow = QsT + (size_t)(qbase + m) * F_DIM;
#pragma unroll
  for (int fc = 0; fc < 8; ++fc) Afrag[fc] = load_a(qrow + fc * 32, half);

  v16bf onesB;   // all-ones B: one extra WMMA computes the row-sum tile
#pragma unroll
  for (int i = 0; i < 16; ++i) onesB[i] = (bf16)1.0f;

  v8f O0 = {}, O1 = {}, O2 = {}, O3 = {}, Lacc = {};
  float mrow[8];
#pragma unroll
  for (int v = 0; v < 8; ++v) mrow[v] = -1e30f;

  stage(0, 0);
  for (int ch = 0; ch < NCHUNK; ++ch) {
    const int buf = ch & 1;
    wait_async0();
    __syncthreads();               // chunk `ch` resident; prior reads retired
    if (ch + 1 < NCHUNK) stage((ch + 1) * 32, buf ^ 1);

    // ---- S = Qs_tile . K_tile (two 16x16 tiles, K-dim = 256), from LDS --
    v8f S0 = {}, S1 = {};
    const bf16* kr0 = &Ks[buf][(size_t)m * F_DIM];
    const bf16* kr1 = &Ks[buf][(size_t)(16 + m) * F_DIM];
#pragma unroll
    for (int fc = 0; fc < 8; ++fc)
      S0 = wmma_bf16(Afrag[fc], load_b(kr0 + fc * 32, half), S0);
#pragma unroll
    for (int fc = 0; fc < 8; ++fc)
      S1 = wmma_bf16(Afrag[fc], load_b(kr1 + fc * 32, half), S1);

    // ---- online softmax: row max via 16-lane xor tree ----
    float scale[8];
#pragma unroll
    for (int v = 0; v < 8; ++v) {
      float t = fmaxf(S0[v], S1[v]);
      t = fmaxf(t, __shfl_xor(t, 1));
      t = fmaxf(t, __shfl_xor(t, 2));
      t = fmaxf(t, __shfl_xor(t, 4));
      t = fmaxf(t, __shfl_xor(t, 8));
      const float nm = fmaxf(mrow[v], t);
      scale[v] = __expf(mrow[v] - nm);
      mrow[v] = nm;
      S0[v] = __expf(S0[v] - nm);
      S1[v] = __expf(S1[v] - nm);
    }
#pragma unroll
    for (int v = 0; v < 8; ++v) {
      O0[v] *= scale[v]; O1[v] *= scale[v]; O2[v] *= scale[v]; O3[v] *= scale[v];
      Lacc[v] *= scale[v];
    }

    // ---- P (D layout) -> LDS -> A fragment (cross-lane relayout) ----
#pragma unroll
    for (int v = 0; v < 8; ++v) {
      const int r = v + 8 * half;
      lds[r * 32 + m]      = (bf16)S0[v];
      lds[r * 32 + 16 + m] = (bf16)S1[v];
    }
    asm volatile("s_wait_dscnt 0x0" ::: "memory");
    FragU pf;
    pf.b[0] = *(const B128*)&lds[m * 32 + half * 8];
    pf.b[1] = *(const B128*)&lds[m * 32 + 16 + half * 8];

    // ---- O += P . V^T (4 c-tiles) and L += P . 1 ----
    const bf16* vr = &Vs[buf][(size_t)m * 32];
    O0 = wmma_bf16(pf.v, load_b(vr + (0 * 16) * 32, half), O0);
    O1 = wmma_bf16(pf.v, load_b(vr + (1 * 16) * 32, half), O1);
    O2 = wmma_bf16(pf.v, load_b(vr + (2 * 16) * 32, half), O2);
    O3 = wmma_bf16(pf.v, load_b(vr + (3 * 16) * 32, half), O3);
    Lacc = wmma_bf16(pf.v, onesB, Lacc);
  }

  // ---- epilogue: unnormalized partials (c-major rows => coalesced) ----
  float* Ob = Opart + ((size_t)split * SP + qbase) * C_DIM;
  v8f* accs[4] = {&O0, &O1, &O2, &O3};
#pragma unroll
  for (int ct = 0; ct < 4; ++ct)
#pragma unroll
    for (int v = 0; v < 8; ++v)
      Ob[(size_t)(v + 8 * half) * C_DIM + ct * 16 + m] = (*accs[ct])[v];
  if (m == 0) {
#pragma unroll
    for (int v = 0; v < 8; ++v) {
      const int r = qbase + v + 8 * half;
      Mpart[split * SP + r] = mrow[v];
      Lpart[split * SP + r] = Lacc[v];
    }
  }
}

// ---- Kernel 5: merge the KSPLIT partials (flash-decoding combine) -------
__global__ __launch_bounds__(256) void combine(
    const float* __restrict__ Opart, const float* __restrict__ Mpart,
    const float* __restrict__ Lpart, float* __restrict__ out) {
  const int t = blockIdx.x * blockDim.x + threadIdx.x;   // over SP*C
  const int c = t & (C_DIM - 1);
  const int q = t >> 6;
  float M = -1e30f;
#pragma unroll
  for (int s = 0; s < KSPLIT; ++s) M = fmaxf(M, Mpart[s * SP + q]);
  float L = 0.f, acc = 0.f;
#pragma unroll
  for (int s = 0; s < KSPLIT; ++s) {
    const float wgt = __expf(Mpart[s * SP + q] - M);
    L += Lpart[s * SP + q] * wgt;
    acc += Opart[((size_t)s * SP + q) * C_DIM + c] * wgt;
  }
  out[(size_t)c * SP + q] = acc / L;
}

extern "C" void kernel_launch(void* const* d_in, const int* in_sizes, int n_in,
                              void* d_out, int out_size, void* d_ws, size_t ws_size,
                              hipStream_t stream) {
  const float* X       = (const float*)d_in[0];
  const float* WQ_task = (const float*)d_in[1];
  const float* BQ_task = (const float*)d_in[2];
  const float* WK_task = (const float*)d_in[3];
  const float* BK_task = (const float*)d_in[4];
  const float* WV_task = (const float*)d_in[5];
  const float* BV_task = (const float*)d_in[6];
  const float* WQ_tm1  = (const float*)d_in[7];
  const float* WQ_x    = (const float*)d_in[8];
  const float* BQ      = (const float*)d_in[9];
  const float* WK_x    = (const float*)d_in[10];
  const float* BK      = (const float*)d_in[11];
  const float* prevQ   = (const float*)d_in[12];
  const float* Vconv_w = (const float*)d_in[13];
  const float* Vconv_b = (const float*)d_in[14];

  char* ws = (char*)d_ws;
  const size_t SZT  = (size_t)SP * F_DIM * sizeof(bf16);       // 4,718,592
  const size_t SZXP = (size_t)PW * PW * F_DIM * sizeof(bf16);  // 4,917,248
  bf16* QsT   = (bf16*)(ws);
  bf16* KT    = (bf16*)(ws + SZT);
  bf16* XTpad = (bf16*)(ws + 2 * SZT);
  bf16* Vbuf  = (bf16*)(ws + 2 * SZT + SZXP);
  bf16* Wc    = (bf16*)(ws + 2 * SZT + SZXP + (size_t)C_DIM * SP * sizeof(bf16));
  char* ws2   = ws + 2 * SZT + SZXP + (size_t)C_DIM * SP * sizeof(bf16) +
                (size_t)9 * C_DIM * F_DIM * sizeof(bf16);
  float* Opart = (float*)(ws2);
  float* Mpart = (float*)(ws2 + (size_t)KSPLIT * SP * C_DIM * sizeof(float));
  float* Lpart = (float*)(ws2 + (size_t)KSPLIT * SP * C_DIM * sizeof(float) +
                          (size_t)KSPLIT * SP * sizeof(float));

  hipMemsetAsync(XTpad, 0, SZXP, stream);   // zero halo (capture-safe)
  qk_prep<<<dim3(F_DIM, SP / 256), 256, 0, stream>>>(
      X, WQ_task, BQ_task, WK_task, BK_task, WQ_tm1, WQ_x, BQ, WK_x, BK, prevQ,
      QsT, KT, XTpad);
  w_prep<<<dim3((9 * C_DIM * F_DIM) / 256), 256, 0, stream>>>(Vconv_w, Wc);
  conv_v<<<dim3(SP / 16 / 4), 128, 0, stream>>>(XTpad, Wc, WV_task, BV_task,
                                                Vconv_b, Vbuf);
  attn<<<dim3(SP / 16 / AT_WAVES, KSPLIT), AT_WAVES * 32, 0, stream>>>(
      QsT, KT, Vbuf, Opart, Mpart, Lpart);
  combine<<<dim3(SP * C_DIM / 256), 256, 0, stream>>>(Opart, Mpart, Lpart,
                                                      (float*)d_out);
}
